// TemporalFlowNetwork_52080773431616
// MI455X (gfx1250) — matrix operand order repaired and checked by reading, not compiled, not run on hardware
//
#include <hip/hip_runtime.h>
#include <hip/hip_bf16.h>

// ---------------------------------------------------------------------------
// Types for CDNA5 WMMA (wave32, 16x16x32 bf16 -> f32)
// ---------------------------------------------------------------------------
typedef __bf16 bf16_t;
typedef __attribute__((ext_vector_type(16))) __bf16 v16bf;
typedef __attribute__((ext_vector_type(8)))  __bf16 v8bf;
typedef __attribute__((ext_vector_type(8)))  float  v8f;

static constexpr int   TFN_L    = 4;
static constexpr int   TFN_H    = 256;
static constexpr int   TFN_S    = 128;
static constexpr int   TFN_IN   = 8;
static constexpr int   TFN_B    = 64;
static constexpr int   TFN_SEQ  = 2048;
static constexpr long  TFN_T    = (long)TFN_B * TFN_SEQ;  // 131072 tokens
static constexpr float TFN_EPS  = 1e-5f;

__device__ __forceinline__ float sigm_f(float v) { return 1.0f / (1.0f + __expf(-v)); }

__device__ __forceinline__ v8f wmma_bf16(v16bf a, v16bf b, v8f c) {
    return __builtin_amdgcn_wmma_f32_16x16x32_bf16(false, a, false, b, (short)0, c, false, false);
}

// A fragment: 16x32 bf16, row-major source. lane = m | (hk<<4); elements 0..15
// hold K = k0 + hk*16 .. +15.
__device__ __forceinline__ v16bf load_a_frag(const bf16_t* __restrict__ A, int lda,
                                             int m0, int k0, int lane) {
    int m = lane & 15, hk = lane >> 4;
    const bf16_t* p = A + (size_t)(m0 + m) * lda + k0 + hk * 16;
    v16bf a;
    *reinterpret_cast<v8bf*>(&a)       = *reinterpret_cast<const v8bf*>(p);
    *(reinterpret_cast<v8bf*>(&a) + 1) = *reinterpret_cast<const v8bf*>(p + 8);
    return a;
}

// B fragment: weight stored N x K row-major (pre-transposed); lane n reads a
// contiguous run of 16 K-values.
__device__ __forceinline__ v16bf load_b_frag(const bf16_t* __restrict__ Wt, int Kd,
                                             int n0, int k0, int lane) {
    int n = lane & 15, hk = lane >> 4;
    return *reinterpret_cast<const v16bf*>(Wt + (size_t)(n0 + n) * Kd + k0 + hk * 16);
}

// ---------------------------------------------------------------------------
// Weight prep kernels (f32 -> bf16, optional transpose / tanh)
// ---------------------------------------------------------------------------
__global__ void cvt_bf16_k(bf16_t* __restrict__ dst, const float* __restrict__ src, int n) {
    int i = blockIdx.x * blockDim.x + threadIdx.x;
    if (i < n) dst[i] = (bf16_t)src[i];
}
__global__ void tanh_bf16_k(bf16_t* __restrict__ dst, const float* __restrict__ src, int n) {
    int i = blockIdx.x * blockDim.x + threadIdx.x;
    if (i < n) dst[i] = (bf16_t)tanhf(src[i]);
}
// src is (K,N) row-major; dst is (N,K) row-major bf16.
__global__ void tpose_bf16_k(bf16_t* __restrict__ dst, const float* __restrict__ src,
                             int K, int N) {
    int i = blockIdx.x * blockDim.x + threadIdx.x;
    if (i >= K * N) return;
    int n = i / K, k = i - n * K;
    dst[i] = (bf16_t)src[(size_t)k * N + n];
}

// ---------------------------------------------------------------------------
// Input projection: h = x @ ip_W + ip_b   (K=8, scalar VALU)
// ---------------------------------------------------------------------------
__global__ __launch_bounds__(256) void input_k(const float* __restrict__ x,
                                               const float* __restrict__ W,
                                               const float* __restrict__ b,
                                               float* __restrict__ h) {
    long tok = blockIdx.x;
    int  c   = threadIdx.x;
    const float* xr = x + tok * TFN_IN;
    float s = b[c];
#pragma unroll
    for (int i = 0; i < TFN_IN; ++i) s += xr[i] * W[i * TFN_H + c];
    h[tok * TFN_H + c] = s;
}

// ---------------------------------------------------------------------------
// Basis kernel (per token): basis = softmax(silu(x@W1+b1)@W2+b2);
// h += basis @ coeffs ; writes h (f32) and hb (bf16).
// ---------------------------------------------------------------------------
__global__ __launch_bounds__(256) void basis_k(const float* __restrict__ x,
                                               const float* __restrict__ W1,
                                               const float* __restrict__ b1,
                                               const float* __restrict__ W2,
                                               const float* __restrict__ b2,
                                               const float* __restrict__ coef,
                                               float* __restrict__ h,
                                               bf16_t* __restrict__ hb) {
    long tok = blockIdx.x;
    int  tid = threadIdx.x;
    __shared__ float t1[TFN_H];
    __shared__ float part[256];
    __shared__ float acc[8];
    __shared__ float sb[8];

    const float* xr = x + tok * TFN_IN;
    float s = b1[tid];
#pragma unroll
    for (int i = 0; i < TFN_IN; ++i) s += xr[i] * W1[i * TFN_H + tid];
    t1[tid] = s * sigm_f(s);  // silu
    __syncthreads();

    int nn = tid & 7, j0 = tid >> 3;  // 8 outputs x 32 partials
    float p = 0.f;
#pragma unroll
    for (int k = 0; k < 8; ++k) {
        int j = j0 + 32 * k;
        p += t1[j] * W2[j * 8 + nn];
    }
    part[nn * 32 + j0] = p;
    __syncthreads();
    if (tid < 8) {
        float a = b2[tid];
        for (int i = 0; i < 32; ++i) a += part[tid * 32 + i];
        acc[tid] = a;
    }
    __syncthreads();
    if (tid == 0) {
        float mx = acc[0];
        for (int n = 1; n < 8; ++n) mx = fmaxf(mx, acc[n]);
        float e[8], sum = 0.f;
        for (int n = 0; n < 8; ++n) { e[n] = __expf(acc[n] - mx); sum += e[n]; }
        for (int n = 0; n < 8; ++n) sb[n] = e[n] / sum;
    }
    __syncthreads();

    size_t idx = (size_t)tok * TFN_H + tid;
    float  hv  = h[idx];
#pragma unroll
    for (int n = 0; n < 8; ++n) hv += sb[n] * coef[n * TFN_H + tid];
    h[idx]  = hv;
    hb[idx] = (bf16_t)hv;
}

// ---------------------------------------------------------------------------
// Generic WMMA GEMM with fused epilogues.
//   EPI 0: u-store   : out f32 to u[(t*64+b)*128+col]  (p0=u)
//   EPI 1: silu->bf16: p1[row*NTOT+col]
//   EPI 2: gate-mix  : g=sigm(v); p0[idx]=g*p0+(1-g)*p2; p1=bf16(p0) (N=256)
//   EPI 3: LN+silu   : row LayerNorm over NTOT=512 (g=p2,b=p3) -> bf16 p1
//   EPI 4: tr-norm   : h=(p2+tr)/(1+0.1*||tr||), NTOT=256 -> p0 f32, p1 bf16
// Block = 256 threads = 8 waves; block covers 16 rows x NTOT cols; wave w owns
// tiles at columns (w + 8*j)*16, j < NT.
// ---------------------------------------------------------------------------
template <int EPI, int NT, int NTOT>
__global__ __launch_bounds__(256) void gemm_k(const bf16_t* __restrict__ A, int lda, int K,
                                              const bf16_t* __restrict__ Wt,
                                              const float* __restrict__ bias,
                                              float* __restrict__ p0,
                                              bf16_t* __restrict__ p1,
                                              const float* __restrict__ p2,
                                              const float* __restrict__ p3) {
    int tid  = threadIdx.x;
    int lane = tid & 31;
    int w    = tid >> 5;
    int m0   = blockIdx.x * 16;
    int n    = lane & 15, hk = lane >> 4;

    __shared__ float smem[(EPI >= 3) ? 16 * NTOT : 16];
    __shared__ float red[(EPI >= 3) ? 544 : 16];

    v8f c[NT];
#pragma unroll
    for (int j = 0; j < NT; ++j) {
        int   col0 = (w + 8 * j) * 16;
        float bv   = bias ? bias[col0 + n] : 0.0f;
#pragma unroll
        for (int r = 0; r < 8; ++r) c[j][r] = bv;
    }

    for (int k0 = 0; k0 < K; k0 += 32) {
        __builtin_prefetch(A + (size_t)(m0 + n) * lda + k0 + 64, 0, 1);
        v16bf a = load_a_frag(A, lda, m0, k0, lane);
#pragma unroll
        for (int j = 0; j < NT; ++j) {
            v16bf b = load_b_frag(Wt, K, (w + 8 * j) * 16, k0, lane);
            c[j]    = wmma_bf16(a, b, c[j]);
        }
    }

    if constexpr (EPI <= 2) {
#pragma unroll
        for (int j = 0; j < NT; ++j) {
#pragma unroll
            for (int r = 0; r < 8; ++r) {
                int   row = m0 + r + 8 * hk;
                int   col = (w + 8 * j) * 16 + n;
                float v   = c[j][r];
                if constexpr (EPI == 0) {  // u, transposed to (t, b) layout
                    int t = row & (TFN_SEQ - 1);
                    int b = row >> 11;
                    p0[((size_t)t * TFN_B + b) * TFN_S + col] = v;
                } else if constexpr (EPI == 1) {
                    p1[(size_t)row * NTOT + col] = (bf16_t)(v * sigm_f(v));
                } else {  // EPI == 2: gate mix
                    float  g   = sigm_f(v);
                    size_t idx = (size_t)row * NTOT + col;
                    float  res = g * p0[idx] + (1.0f - g) * p2[idx];
                    p0[idx] = res;
                    p1[idx] = (bf16_t)res;
                }
            }
        }
    } else {
        // stash accumulators into LDS for row-wise reductions
#pragma unroll
        for (int j = 0; j < NT; ++j)
#pragma unroll
            for (int r = 0; r < 8; ++r)
                smem[(r + 8 * hk) * NTOT + (w + 8 * j) * 16 + n] = c[j][r];
        __syncthreads();

        int       row = tid >> 4, sub = tid & 15;
        const int per = NTOT / 16;
        float     s = 0.f, q = 0.f;
        for (int i = 0; i < per; ++i) {
            float v = smem[row * NTOT + sub * per + i];
            s += v;
            q += v * v;
        }
        red[row * 16 + sub]       = s;
        red[256 + row * 16 + sub] = q;
        __syncthreads();
        if (sub == 0) {
            float S = 0.f, Q = 0.f;
            for (int i = 0; i < 16; ++i) {
                S += red[row * 16 + i];
                Q += red[256 + row * 16 + i];
            }
            if constexpr (EPI == 3) {
                float mean = S / (float)NTOT;
                red[512 + row] = mean;
                red[528 + row] = rsqrtf(Q / (float)NTOT - mean * mean + TFN_EPS);
            } else {  // EPI == 4: ||tr|| over the row
                red[512 + row] = 1.0f / (1.0f + 0.1f * sqrtf(Q));
            }
        }
        __syncthreads();
        if constexpr (EPI == 3) {
            float mean = red[512 + row], inv = red[528 + row];
            for (int i = 0; i < per; ++i) {
                int   col = sub * per + i;
                float v   = (smem[row * NTOT + col] - mean) * inv * p2[col] + p3[col];
                p1[(size_t)(m0 + row) * NTOT + col] = (bf16_t)(v * sigm_f(v));
            }
        } else {
            float scale = red[512 + row];
            for (int i = 0; i < per; ++i) {
                int    col = sub * per + i;
                size_t idx = (size_t)(m0 + row) * NTOT + col;
                float  hn  = (p2[idx] + smem[row * NTOT + col]) * scale;
                p0[idx] = hn;
                p1[idx] = (bf16_t)hn;
            }
        }
    }
}

// ---------------------------------------------------------------------------
// Dual-source GEMM: out = statesb(Tx128)@Cw(128x256) + hb(Tx256)@Dw(256x256)
// ---------------------------------------------------------------------------
__global__ __launch_bounds__(256) void gemm_dual_k(const bf16_t* __restrict__ As,
                                                   const bf16_t* __restrict__ Cw,
                                                   const bf16_t* __restrict__ Ah,
                                                   const bf16_t* __restrict__ Dw,
                                                   float* __restrict__ out) {
    int tid  = threadIdx.x;
    int lane = tid & 31;
    int w    = tid >> 5;
    int m0   = blockIdx.x * 16;
    int n    = lane & 15, hk = lane >> 4;

    v8f c[2];
#pragma unroll
    for (int j = 0; j < 2; ++j)
#pragma unroll
        for (int r = 0; r < 8; ++r) c[j][r] = 0.0f;

    for (int k0 = 0; k0 < TFN_S; k0 += 32) {
        v16bf a = load_a_frag(As, TFN_S, m0, k0, lane);
#pragma unroll
        for (int j = 0; j < 2; ++j) {
            v16bf b = load_b_frag(Cw, TFN_S, (w + 8 * j) * 16, k0, lane);
            c[j]    = wmma_bf16(a, b, c[j]);
        }
    }
    for (int k0 = 0; k0 < TFN_H; k0 += 32) {
        __builtin_prefetch(Ah + (size_t)(m0 + n) * TFN_H + k0 + 64, 0, 1);
        v16bf a = load_a_frag(Ah, TFN_H, m0, k0, lane);
#pragma unroll
        for (int j = 0; j < 2; ++j) {
            v16bf b = load_b_frag(Dw, TFN_H, (w + 8 * j) * 16, k0, lane);
            c[j]    = wmma_bf16(a, b, c[j]);
        }
    }
#pragma unroll
    for (int j = 0; j < 2; ++j)
#pragma unroll
        for (int r = 0; r < 8; ++r) {
            int row = m0 + r + 8 * hk;
            int col = (w + 8 * j) * 16 + n;
            out[(size_t)row * TFN_H + col] = c[j][r];
        }
}

// ---------------------------------------------------------------------------
// Sequential scan: S(64x128) <- S @ tanh(A)^T + U_t, WMMA recurrence.
// One workgroup; state kept in LDS (bf16); Al B-fragments preloaded in VGPRs.
// u laid out (t, b, s); states emitted bf16 at (b*SEQ+t, s) for the next GEMM.
// ---------------------------------------------------------------------------
__global__ __launch_bounds__(256) void scan_k(const bf16_t* __restrict__ Alb,
                                              const float* __restrict__ u,
                                              bf16_t* __restrict__ statesb) {
    __shared__ bf16_t sst[TFN_B * TFN_S];  // 64 x 128 bf16 = 16 KB
    int tid  = threadIdx.x;
    int lane = tid & 31;
    int w    = tid >> 5;
    int n    = lane & 15, hk = lane >> 4;
    int n0   = w * 16;

    for (int i = tid; i < TFN_B * TFN_S; i += 256) sst[i] = (bf16_t)0.0f;
    __syncthreads();

    v16bf bf[4];  // constant tanh(A)^T fragments for this wave's 16 columns
#pragma unroll
    for (int kc = 0; kc < 4; ++kc)
        bf[kc] = *reinterpret_cast<const v16bf*>(Alb + (size_t)(n0 + n) * TFN_S +
                                                 kc * 32 + hk * 16);

    for (int t = 0; t < TFN_SEQ; ++t) {
        v8f c[4];
#pragma unroll
        for (int mt = 0; mt < 4; ++mt) {
#pragma unroll
            for (int r = 0; r < 8; ++r) {
                int b   = mt * 16 + r + 8 * hk;
                c[mt][r] = u[((size_t)t * TFN_B + b) * TFN_S + n0 + n];
            }
#pragma unroll
            for (int kc = 0; kc < 4; ++kc) {
                v16bf a;
                const bf16_t* p = &sst[(mt * 16 + n) * TFN_S + kc * 32 + hk * 16];
                *reinterpret_cast<v8bf*>(&a)       = *reinterpret_cast<const v8bf*>(p);
                *(reinterpret_cast<v8bf*>(&a) + 1) = *reinterpret_cast<const v8bf*>(p + 8);
                c[mt] = wmma_bf16(a, bf[kc], c[mt]);
            }
        }
        __syncthreads();  // all reads of previous state complete
#pragma unroll
        for (int mt = 0; mt < 4; ++mt)
#pragma unroll
            for (int r = 0; r < 8; ++r) {
                int    b = mt * 16 + r + 8 * hk;
                bf16_t v = (bf16_t)c[mt][r];
                sst[b * TFN_S + n0 + n] = v;
                statesb[((size_t)b * TFN_SEQ + t) * TFN_S + n0 + n] = v;
            }
        __syncthreads();  // publish new state
    }
}

// ---------------------------------------------------------------------------
// Head: final = h[:, -1, :]; two small MLP layers + LN; 64 rows.
// ---------------------------------------------------------------------------
__global__ __launch_bounds__(256) void head_k(const float* __restrict__ h,
                                              const float* __restrict__ W1,
                                              const float* __restrict__ b1,
                                              const float* __restrict__ lg,
                                              const float* __restrict__ lb,
                                              const float* __restrict__ W2,
                                              const float* __restrict__ b2,
                                              const float* __restrict__ W3,
                                              const float* __restrict__ b3,
                                              float* __restrict__ out) {
    int b   = blockIdx.x;
    int tid = threadIdx.x;
    __shared__ float f[256];
    __shared__ float z1[512];
    __shared__ float z2[256];
    __shared__ float ps[256], pq[256];
    __shared__ float stat[2];

    f[tid] = h[((size_t)b * TFN_SEQ + (TFN_SEQ - 1)) * TFN_H + tid];
    __syncthreads();

    float y[2];
#pragma unroll
    for (int c2 = 0; c2 < 2; ++c2) {
        int   c = tid * 2 + c2;
        float s = b1[c];
        for (int j = 0; j < 256; ++j) s += f[j] * W1[j * 512 + c];
        y[c2] = s;
    }
    ps[tid] = y[0] + y[1];
    pq[tid] = y[0] * y[0] + y[1] * y[1];
    __syncthreads();
    if (tid == 0) {
        float S = 0.f, Q = 0.f;
        for (int i = 0; i < 256; ++i) { S += ps[i]; Q += pq[i]; }
        float mean = S / 512.0f;
        stat[0] = mean;
        stat[1] = rsqrtf(Q / 512.0f - mean * mean + TFN_EPS);
    }
    __syncthreads();
#pragma unroll
    for (int c2 = 0; c2 < 2; ++c2) {
        int   c = tid * 2 + c2;
        float v = (y[c2] - stat[0]) * stat[1] * lg[c] + lb[c];
        z1[c]   = v * sigm_f(v);
    }
    __syncthreads();
    float s2 = b2[tid];
    for (int j = 0; j < 512; ++j) s2 += z1[j] * W2[j * 256 + tid];
    z2[tid] = s2 * sigm_f(s2);
    __syncthreads();
    if (tid < 2) {
        float s = b3[tid];
        for (int j = 0; j < 256; ++j) s += z2[j] * W3[j * 2 + tid];
        out[b * 2 + tid] = s;
    }
}

// ---------------------------------------------------------------------------
// Host launcher
// ---------------------------------------------------------------------------
extern "C" void kernel_launch(void* const* d_in, const int* in_sizes, int n_in,
                              void* d_out, int out_size, void* d_ws, size_t ws_size,
                              hipStream_t stream) {
    const float* x     = (const float*)d_in[0];
    const float* ipW   = (const float*)d_in[1];
    const float* ipb   = (const float*)d_in[2];
    const float* bnW1  = (const float*)d_in[3];
    const float* bnb1  = (const float*)d_in[4];
    const float* bnW2  = (const float*)d_in[5];
    const float* bnb2  = (const float*)d_in[6];
    const float* coeffs= (const float*)d_in[7];
    const float* Asrc  = (const float*)d_in[8];
    const float* Bsrc  = (const float*)d_in[9];
    const float* Csrc  = (const float*)d_in[10];
    const float* Dsrc  = (const float*)d_in[11];
    const float* gW1   = (const float*)d_in[12];
    const float* gb1   = (const float*)d_in[13];
    const float* gW2   = (const float*)d_in[14];
    const float* gb2   = (const float*)d_in[15];
    const float* fW1   = (const float*)d_in[16];
    const float* fb1   = (const float*)d_in[17];
    const float* flng  = (const float*)d_in[18];
    const float* flnb  = (const float*)d_in[19];
    const float* fW2   = (const float*)d_in[20];
    const float* fb2   = (const float*)d_in[21];
    const float* sW1   = (const float*)d_in[22];
    const float* sb1   = (const float*)d_in[23];
    const float* slg   = (const float*)d_in[24];
    const float* slb   = (const float*)d_in[25];
    const float* sW2   = (const float*)d_in[26];
    const float* sb2   = (const float*)d_in[27];
    const float* sW3   = (const float*)d_in[28];
    const float* sb3   = (const float*)d_in[29];

    char*  wsp = (char*)d_ws;
    size_t off = 0;
    auto alloc = [&](size_t bytes) -> void* {
        void* p = wsp + off;
        off = (off + bytes + 255) & ~(size_t)255;
        return p;
    };

    const size_t T = (size_t)TFN_T;
    float*  h       = (float*)alloc(T * 256 * 4);
    float*  outbuf  = (float*)alloc(T * 256 * 4);
    float*  u       = (float*)alloc(T * 128 * 4);
    bf16_t* hb      = (bf16_t*)alloc(T * 256 * 2);
    bf16_t* outb    = (bf16_t*)alloc(T * 256 * 2);
    bf16_t* zbuf    = (bf16_t*)alloc(T * 512 * 2);
    bf16_t* statesb = (bf16_t*)alloc(T * 128 * 2);
    bf16_t* Bb      = (bf16_t*)alloc((size_t)TFN_L * 128 * 256 * 2);
    bf16_t* Cb      = (bf16_t*)alloc((size_t)TFN_L * 256 * 128 * 2);
    bf16_t* Alb     = (bf16_t*)alloc((size_t)TFN_L * 128 * 128 * 2);
    bf16_t* Dbt     = (bf16_t*)alloc((size_t)TFN_L * 256 * 256 * 2);
    bf16_t* gW1t    = (bf16_t*)alloc((size_t)TFN_L * 512 * 256 * 2);
    bf16_t* gW2t    = (bf16_t*)alloc((size_t)TFN_L * 256 * 512 * 2);
    bf16_t* fW1t    = (bf16_t*)alloc((size_t)TFN_L * 512 * 256 * 2);
    bf16_t* fW2t    = (bf16_t*)alloc((size_t)TFN_L * 256 * 512 * 2);

    // ---- weight prep (bf16, N x K layout) ----
    auto blk = [](int n) { return dim3((n + 255) / 256); };
    cvt_bf16_k<<<blk(TFN_L * 128 * 256), 256, 0, stream>>>(Bb, Bsrc, TFN_L * 128 * 256);
    cvt_bf16_k<<<blk(TFN_L * 256 * 128), 256, 0, stream>>>(Cb, Csrc, TFN_L * 256 * 128);
    tanh_bf16_k<<<blk(TFN_L * 128 * 128), 256, 0, stream>>>(Alb, Asrc, TFN_L * 128 * 128);
    for (int l = 0; l < TFN_L; ++l) {
        tpose_bf16_k<<<blk(256 * 256), 256, 0, stream>>>(Dbt + (size_t)l * 65536,
                                                         Dsrc + (size_t)l * 65536, 256, 256);
        tpose_bf16_k<<<blk(256 * 512), 256, 0, stream>>>(gW1t + (size_t)l * 131072,
                                                         gW1 + (size_t)l * 131072, 256, 512);
        tpose_bf16_k<<<blk(512 * 256), 256, 0, stream>>>(gW2t + (size_t)l * 131072,
                                                         gW2 + (size_t)l * 131072, 512, 256);
        tpose_bf16_k<<<blk(256 * 512), 256, 0, stream>>>(fW1t + (size_t)l * 131072,
                                                         fW1 + (size_t)l * 131072, 256, 512);
        tpose_bf16_k<<<blk(512 * 256), 256, 0, stream>>>(fW2t + (size_t)l * 131072,
                                                         fW2 + (size_t)l * 131072, 512, 256);
    }

    // ---- forward pass ----
    const int MB = (int)(T / 16);  // 8192 M-strips
    input_k<<<(int)T, 256, 0, stream>>>(x, ipW, ipb, h);

    for (int l = 0; l < TFN_L; ++l) {
        basis_k<<<(int)T, 256, 0, stream>>>(x, bnW1 + l * 2048, bnb1 + l * 256,
                                            bnW2 + l * 2048, bnb2 + l * 8,
                                            coeffs + l * 2048, h, hb);
        // u = h @ B^T  (writes (t,b,s) layout)
        gemm_k<0, 1, 128><<<MB, 256, 0, stream>>>(hb, 256, 256, Bb + (size_t)l * 32768,
                                                  nullptr, u, nullptr, nullptr, nullptr);
        // sequential WMMA recurrence
        scan_k<<<1, 256, 0, stream>>>(Alb + (size_t)l * 16384, u, statesb);
        // out = states @ C^T + h @ D
        gemm_dual_k<<<MB, 256, 0, stream>>>(statesb, Cb + (size_t)l * 32768, hb,
                                            Dbt + (size_t)l * 65536, outbuf);
        // z1 = silu(h @ gW1 + gb1)
        gemm_k<1, 4, 512><<<MB, 256, 0, stream>>>(hb, 256, 256, gW1t + (size_t)l * 131072,
                                                  gb1 + l * 512, nullptr, zbuf, nullptr,
                                                  nullptr);
        // gate mix: out = sigm(z1@gW2+gb2)*out + (1-g)*h -> outbuf, outb
        gemm_k<2, 2, 256><<<MB, 256, 0, stream>>>(zbuf, 512, 512, gW2t + (size_t)l * 131072,
                                                  gb2 + l * 256, outbuf, outb, h, nullptr);
        // tf = silu(LN(out@fW1+fb1))
        gemm_k<3, 4, 512><<<MB, 256, 0, stream>>>(outb, 256, 256, fW1t + (size_t)l * 131072,
                                                  fb1 + l * 512, nullptr, zbuf, flng + l * 512,
                                                  flnb + l * 512);
        // h = (out + tf@fW2+fb2) / (1 + 0.1*||tr||)
        gemm_k<4, 2, 256><<<MB, 256, 0, stream>>>(zbuf, 512, 512, fW2t + (size_t)l * 131072,
                                                  fb2 + l * 256, h, hb, outbuf, nullptr);
    }

    head_k<<<TFN_B, 256, 0, stream>>>(h, sW1, sb1, slg, slb, sW2, sb2, sW3, sb3,
                                      (float*)d_out);
}
